// DeformConv_824633720944
// MI455X (gfx1250) — matrix operand order repaired and checked by reading, not compiled, not run on hardware
//
#include <hip/hip_runtime.h>

// ---------------------------------------------------------------------------
// Deformable Conv (DCNv2-style) for MI455X / gfx1250, wave32 + WMMA bf16.
//   kT      : transpose x (NCHW f32) -> xt (NHWC bf16) so channel gathers are
//             contiguous 16B vector loads (8x fewer VMEM ops in the hot loop)
//   k0a/k0b : reorder conv weights to bf16 [tap][o][c]
//   k1      : offset conv (27ch) as WMMA GEMM -> off_y / off_x / sigmoid(mask)
//   k2      : fused bilinear-gather + bf16 WMMA GEMM + bias + BN + ReLU
// Double-buffered LDS A tiles, software-pipelined build, b128 LDS traffic.
// ---------------------------------------------------------------------------

typedef __bf16   bf16_t;
typedef bf16_t   bf16x16 __attribute__((ext_vector_type(16)));
typedef float    v8f     __attribute__((ext_vector_type(8)));
typedef float    f32x2   __attribute__((ext_vector_type(2)));
typedef unsigned int u32x4 __attribute__((ext_vector_type(4)));

union Frag {            // 32 bytes per lane = one 16x32 bf16 A/B operand slice
  u32x4   u[2];
  bf16x16 v;
};

__device__ __forceinline__ unsigned int bfround(float f) {
  unsigned int u = __float_as_uint(f);
  return u + (0x7fffu + ((u >> 16) & 1u));     // rne, result in high 16 bits
}
__device__ __forceinline__ unsigned int pkbf(float lo, float hi) {
  return (bfround(lo) >> 16) | (bfround(hi) & 0xffff0000u);
}
__device__ __forceinline__ unsigned short f2bf(float f) {
  return (unsigned short)(bfround(f) >> 16);
}
// unpack uint holding two bf16 (ch 2k low, 2k+1 high) -> two f32
__device__ __forceinline__ f32x2 upbf(unsigned int u) {
  f32x2 r;
  r.x = __uint_as_float(u << 16);
  r.y = __uint_as_float(u & 0xffff0000u);
  return r;
}

// ---------------------------------------------------------------------------
// kT: x (8,256,64,64) f32 NCHW -> xt[b][hw][c] bf16 (NHWC). 8192 blocks.
// ---------------------------------------------------------------------------
__global__ __launch_bounds__(256) void k_nhwc(const float* __restrict__ x,
                                              unsigned short* __restrict__ xt) {
  const int bidx = blockIdx.x;            // b(8) x hwTile(128) x cTile(8)
  const int b   = bidx >> 10;
  const int hw0 = ((bidx >> 3) & 127) << 5;
  const int c0  = (bidx & 7) << 5;
  __shared__ unsigned short t[32][33];
  const int tid = threadIdx.x;
  const int rj = tid & 31;                // hw (coalesced read)
  const int ri = tid >> 5;                // c row, 8 apart
  const float* src = x + (((size_t)(b * 256 + c0 + ri)) << 12) + hw0 + rj;
#pragma unroll
  for (int k = 0; k < 4; ++k)
    t[ri + 8 * k][rj] = f2bf(src[((size_t)(8 * k)) << 12]);
  __syncthreads();
  const int wi = (tid & 15) << 1;         // c pair (coalesced write)
  const int wj = tid >> 4;                // hw row, 16 apart
#pragma unroll
  for (int k = 0; k < 2; ++k) {
    const int j = wj + 16 * k;
    unsigned int v = (unsigned int)t[wi][j] | ((unsigned int)t[wi + 1][j] << 16);
    *reinterpret_cast<unsigned int*>(
        xt + (((size_t)((b << 12) + hw0 + j)) << 8) + c0 + wi) = v;
  }
}

// ---------------------------------------------------------------------------
// k0a: weight (O=256, C=256, 3,3) f32  ->  wBt[tap][o][c] bf16   (589824 elems)
// ---------------------------------------------------------------------------
__global__ __launch_bounds__(256) void k_reorder_w(const float* __restrict__ w,
                                                   unsigned short* __restrict__ wBt) {
  int t   = blockIdx.x * 256 + threadIdx.x;   // 2304 blocks
  int tap = t >> 16;
  int rem = t & 65535;
  int o   = rem >> 8;
  int c   = rem & 255;
  wBt[t] = f2bf(w[o * 2304 + c * 9 + tap]);
}

// ---------------------------------------------------------------------------
// k0b: w_off (27, C=256, 3,3) f32 -> wOffT[tap][o(32 padded)][c] bf16 (73728)
// ---------------------------------------------------------------------------
__global__ __launch_bounds__(256) void k_reorder_woff(const float* __restrict__ w,
                                                      unsigned short* __restrict__ wOffT) {
  int t   = blockIdx.x * 256 + threadIdx.x;   // 288 blocks
  int tap = t >> 13;
  int o   = (t >> 8) & 31;
  int c   = t & 255;
  float v = (o < 27) ? w[o * 2304 + c * 9 + tap] : 0.f;
  wOffT[t] = f2bf(v);
}

// ---------------------------------------------------------------------------
// k1: offset conv. grid = 512 (b*64+h). M=64 pixels, N=32, K=2304 (72x K=32).
//     8 waves: wave -> (msub = wave&3, nsub = wave>>2), one 16x16 tile each.
// ---------------------------------------------------------------------------
__global__ __launch_bounds__(256) void k_offset_conv(
    const unsigned short* __restrict__ xt, const unsigned short* __restrict__ wOffT,
    const float* __restrict__ b_off,
    float* __restrict__ offy, float* __restrict__ offx, float* __restrict__ maskS)
{
  __shared__ unsigned short sA[2][64 * 40];   // double buffer, pitch 40 halves

  const int tid  = threadIdx.x;
  const int lane = tid & 31;
  const int wave = tid >> 5;
  const int b    = blockIdx.x >> 6;
  const int h    = blockIdx.x & 63;
  const int msub = wave & 3;
  const int nsub = wave >> 2;

  const int p    = tid & 63;                  // pixel column for A-build
  const int cgrp = tid >> 6;                  // 0..3 -> 8 channels each

  v8f acc = (v8f){0.f, 0.f, 0.f, 0.f, 0.f, 0.f, 0.f, 0.f};

  const int alo    = (lane < 16) ? 0 : 16;    // A frag K base per lane
  const int arowoff= (msub * 16 + (lane & 15)) * 40 + alo;
  const int kkb    = (lane < 16) ? 0 : 16;    // B frag K base per lane
  const int ob     = nsub * 16 + (lane & 15); // B frag column per lane

  const unsigned short* xtb = xt + (((size_t)b) << 20) + cgrp * 8;
  const int wrOff  = p * 40 + cgrp * 8;       // A-tile write slot (b128)

  // branch-free border: clamped NHWC b128 load + all-ones/zero mask
  auto build = [&](int s, int buf) {
    const int tap = s >> 3;
    const int c0  = (s & 7) << 5;
    const int ky  = tap / 3, kx = tap - ky * 3;
    const int yy  = h - 1 + ky;
    const int xc  = p - 1 + kx;
    const unsigned int msk =
        ((yy >= 0) & (yy < 64) & (xc >= 0) & (xc < 64)) ? 0xffffffffu : 0u;
    const int yc  = min(max(yy, 0), 63);
    const int xcc = min(max(xc, 0), 63);
    u32x4 d = *reinterpret_cast<const u32x4*>(xtb + (((yc << 6) + xcc) << 8) + c0);
    d = d & (u32x4){msk, msk, msk, msk};
    *reinterpret_cast<u32x4*>(&sA[buf][wrOff]) = d;    // one ds_store_b128
  };

  build(0, 0);
  for (int step = 0; step < 72; ++step) {
    __syncthreads();
    const int buf = step & 1;
    const int tap = step >> 3;
    const int c0  = (step & 7) << 5;

    Frag fa, fb;
    const unsigned short* ap = &sA[buf][arowoff];
    fa.u[0] = reinterpret_cast<const u32x4*>(ap)[0];   // K {0-7 | 16-23}
    fa.u[1] = reinterpret_cast<const u32x4*>(ap)[1];   // K {8-15 | 24-31}
    const unsigned short* bp = wOffT + ((tap * 32 + ob) << 8) + c0 + kkb;
    fb.u[0] = reinterpret_cast<const u32x4*>(bp)[0];
    fb.u[1] = reinterpret_cast<const u32x4*>(bp)[1];

    if (step < 71) build(step + 1, buf ^ 1);           // pipeline next tile

    acc = __builtin_amdgcn_wmma_f32_16x16x32_bf16(false, fa.v, false, fb.v,
                                                  (short)0, acc, false, false);
  }

  // Epilogue: scatter off_y / off_x / sigmoid(mask) to workspace.
  const int o = nsub * 16 + (lane & 15);
  if (o < 27) {
    const float bo  = b_off[o];
    const int   whi = (lane >= 16) ? 8 : 0;
#pragma unroll
    for (int r = 0; r < 8; ++r) {
      const int wpix = msub * 16 + r + whi;
      float val = acc[r] + bo;
      if (o < 9) {
        offy[((((b * 9 + o) * 64 + h)) << 6) + wpix] = val;
      } else if (o < 18) {
        offx[((((b * 9 + (o - 9)) * 64 + h)) << 6) + wpix] = val;
      } else {
        maskS[((((b * 9 + (o - 18)) * 64 + h)) << 6) + wpix] =
            1.f / (1.f + __expf(-val));
      }
    }
  }
}

// ---------------------------------------------------------------------------
// k2: fused deformable GEMM. grid = 512 (b*64+h). M=64 pixels, N=256, K=2304.
//     8 waves, each owns N=32 (2 cols) x 4 M-subtiles -> 64 acc VGPRs/wave.
// ---------------------------------------------------------------------------
__global__ __launch_bounds__(256) void k_deform_gemm(
    const unsigned short* __restrict__ xt, const unsigned short* __restrict__ wBt,
    const float* __restrict__ offy, const float* __restrict__ offx,
    const float* __restrict__ maskS,
    const float* __restrict__ bias, const float* __restrict__ gamma,
    const float* __restrict__ beta, const float* __restrict__ rmean,
    const float* __restrict__ rvar,
    float* __restrict__ out)
{
  __shared__ unsigned short sA[2][64 * 40];   // double buffer A tiles
  __shared__ unsigned int   sYX[576];         // packed clamped coords (tap,p)
  __shared__ float          sW[576 * 4];      // bilinear wts * mask, validity folded

  const int tid  = threadIdx.x;
  const int lane = tid & 31;
  const int wave = tid >> 5;
  const int b    = blockIdx.x >> 6;
  const int h    = blockIdx.x & 63;

  // ---- precompute sampling params for 9 taps x 64 pixels ----
  for (int i = tid; i < 576; i += 256) {
    const int tap = i >> 6;
    const int pw  = i & 63;
    const int ky  = tap / 3, kx = tap - ky * 3;
    const int base = ((((b * 9 + tap) * 64 + h)) << 6) + pw;
    const float py = (float)(h - 1 + ky) + offy[base];
    const float px = (float)(pw - 1 + kx) + offx[base];
    const float m  = maskS[base];
    const float fy = floorf(py), fx = floorf(px);
    const float wy1 = py - fy, wx1 = px - fx;
    const int y0 = (int)fy, x0 = (int)fx;
    const int y1 = y0 + 1, x1 = x0 + 1;
    const bool vy0 = (y0 >= 0) & (y0 < 64);
    const bool vy1 = (y1 >= 0) & (y1 < 64);
    const bool vx0 = (x0 >= 0) & (x0 < 64);
    const bool vx1 = (x1 >= 0) & (x1 < 64);
    const float wy0 = 1.f - wy1, wx0 = 1.f - wx1;
    sW[i * 4 + 0] = (vy0 && vx0) ? wy0 * wx0 * m : 0.f;
    sW[i * 4 + 1] = (vy0 && vx1) ? wy0 * wx1 * m : 0.f;
    sW[i * 4 + 2] = (vy1 && vx0) ? wy1 * wx0 * m : 0.f;
    sW[i * 4 + 3] = (vy1 && vx1) ? wy1 * wx1 * m : 0.f;
    const unsigned int y0c = (unsigned)min(max(y0, 0), 63);
    const unsigned int y1c = (unsigned)min(max(y1, 0), 63);
    const unsigned int x0c = (unsigned)min(max(x0, 0), 63);
    const unsigned int x1c = (unsigned)min(max(x1, 0), 63);
    sYX[i] = y0c | (y1c << 8) | (x0c << 16) | (x1c << 24);
  }

  v8f acc[4][2];
#pragma unroll
  for (int i = 0; i < 4; ++i)
#pragma unroll
    for (int j = 0; j < 2; ++j)
      acc[i][j] = (v8f){0.f, 0.f, 0.f, 0.f, 0.f, 0.f, 0.f, 0.f};

  const int p     = tid & 63;
  const int cgrp  = tid >> 6;
  const int obase = wave << 5;
  const int alo   = (lane < 16) ? 0 : 16;
  const int arowoff = (lane & 15) * 40 + alo;
  const int kkb   = (lane < 16) ? 0 : 16;
  const int ocol  = obase + (lane & 15);
  const int wrOff = p * 40 + cgrp * 8;
  const unsigned short* xtb = xt + (((size_t)b) << 20) + cgrp * 8;

  // per-thread sampling state, refreshed once per tap (uniform switch)
  int   cur_tap = -1;
  float tw0 = 0.f, tw1 = 0.f, tw2 = 0.f, tw3 = 0.f;
  int   c00 = 0, c01 = 0, c10 = 0, c11 = 0;   // NHWC element offsets (ushorts)

  auto build = [&](int s, int buf) {
    const int tap = s >> 3;
    const int c0  = (s & 7) << 5;
    if (tap != cur_tap) {
      cur_tap = tap;
      const int i = (tap << 6) + p;
      const unsigned int yx = sYX[i];
      tw0 = sW[i * 4 + 0]; tw1 = sW[i * 4 + 1];
      tw2 = sW[i * 4 + 2]; tw3 = sW[i * 4 + 3];
      const int y0o = (int)(yx & 255) << 6;
      const int y1o = (int)((yx >> 8) & 255) << 6;
      const int x0c = (int)((yx >> 16) & 255);
      const int x1c = (int)(yx >> 24);
      c00 = (y0o + x0c) << 8; c01 = (y0o + x1c) << 8;
      c10 = (y1o + x0c) << 8; c11 = (y1o + x1c) << 8;
    }
    const unsigned short* bp = xtb + c0;       // +8 bf16 channels per thread
    const u32x4 qa = *reinterpret_cast<const u32x4*>(bp + c00);
    const u32x4 qb = *reinterpret_cast<const u32x4*>(bp + c01);
    const u32x4 qc = *reinterpret_cast<const u32x4*>(bp + c10);
    const u32x4 qd = *reinterpret_cast<const u32x4*>(bp + c11);
    u32x4 pk;
#pragma unroll
    for (int k = 0; k < 4; ++k) {
      f32x2 r = upbf(qa[k]) * tw0;
      r += upbf(qb[k]) * tw1;
      r += upbf(qc[k]) * tw2;
      r += upbf(qd[k]) * tw3;
      pk[k] = pkbf(r.x, r.y);
    }
    *reinterpret_cast<u32x4*>(&sA[buf][wrOff]) = pk;   // one ds_store_b128
  };

  __syncthreads();            // sYX/sW ready
  build(0, 0);

  for (int step = 0; step < 72; ++step) {
    __syncthreads();
    const int buf = step & 1;
    const int tap = step >> 3;
    const int c0  = (step & 7) << 5;

    // ---- B fragments: 2 x 16 columns, contiguous bf16 from wBt ----
    Frag fb0, fb1;
    {
      const unsigned short* bp0 = wBt + (((tap << 8) + ocol) << 8) + c0 + kkb;
      const unsigned short* bp1 = bp0 + (16 << 8);   // +16 output channels
      fb0.u[0] = reinterpret_cast<const u32x4*>(bp0)[0];
      fb0.u[1] = reinterpret_cast<const u32x4*>(bp0)[1];
      fb1.u[0] = reinterpret_cast<const u32x4*>(bp1)[0];
      fb1.u[1] = reinterpret_cast<const u32x4*>(bp1)[1];
    }

    // ---- A fragments for all 4 M-subtiles ----
    Frag fa[4];
#pragma unroll
    for (int msub = 0; msub < 4; ++msub) {
      const unsigned short* ap = &sA[buf][msub * (16 * 40) + arowoff];
      fa[msub].u[0] = reinterpret_cast<const u32x4*>(ap)[0];
      fa[msub].u[1] = reinterpret_cast<const u32x4*>(ap)[1];
    }

    if (step < 71) build(step + 1, buf ^ 1);           // pipeline next tile

#pragma unroll
    for (int msub = 0; msub < 4; ++msub) {
      acc[msub][0] = __builtin_amdgcn_wmma_f32_16x16x32_bf16(
          false, fa[msub].v, false, fb0.v, (short)0, acc[msub][0], false, false);
      acc[msub][1] = __builtin_amdgcn_wmma_f32_16x16x32_bf16(
          false, fa[msub].v, false, fb1.v, (short)0, acc[msub][1], false, false);
    }
  }

  // ---- epilogue: bias + folded BN + ReLU ----
  const int whi = (lane >= 16) ? 8 : 0;
#pragma unroll
  for (int nsub = 0; nsub < 2; ++nsub) {
    const int o   = obase + nsub * 16 + (lane & 15);
    const float inv   = gamma[o] * rsqrtf(rvar[o] + 1e-5f);
    const float shift = beta[o] - rmean[o] * inv;
    const float bi    = bias[o];
    float* op = out + (((size_t)(b * 256 + o) * 64 + h) << 6);
#pragma unroll
    for (int msub = 0; msub < 4; ++msub) {
#pragma unroll
      for (int r = 0; r < 8; ++r) {
        const int wpix = msub * 16 + r + whi;
        float val = (acc[msub][nsub][r] + bi) * inv + shift;
        op[wpix] = fmaxf(val, 0.f);
      }
    }
  }
}

// ---------------------------------------------------------------------------
// Host-side launcher. Workspace layout (bytes, all 16B aligned):
//   [0, 16777216)             xt    bf16  NHWC (8*4096*256)
//   [16777216, 17956864)      wBt   bf16  (9*256*256)
//   [17956864, 18104320)      wOffT bf16  (9*32*256)
//   [18104320, 19283968)      offy  f32   (8*9*64*64)
//   [19283968, 20463616)      offx  f32
//   [20463616, 21643264)      maskS f32   (sigmoid already applied)
// ---------------------------------------------------------------------------
extern "C" void kernel_launch(void* const* d_in, const int* in_sizes, int n_in,
                              void* d_out, int out_size, void* d_ws, size_t ws_size,
                              hipStream_t stream) {
  const float* x      = (const float*)d_in[0];
  const float* w_off  = (const float*)d_in[1];
  const float* b_off  = (const float*)d_in[2];
  const float* weight = (const float*)d_in[3];
  const float* bias   = (const float*)d_in[4];
  const float* gamma  = (const float*)d_in[5];
  const float* beta   = (const float*)d_in[6];
  const float* rmean  = (const float*)d_in[7];
  const float* rvar   = (const float*)d_in[8];
  float* out = (float*)d_out;

  unsigned short* xt    = (unsigned short*)d_ws;            // 8388608 ushorts
  unsigned short* wBt   = xt + 8388608;
  unsigned short* wOffT = wBt + 589824;
  float* offy  = (float*)(wOffT + 73728);
  float* offx  = offy + 294912;
  float* maskS = offx + 294912;

  k_nhwc        <<<8192, 256, 0, stream>>>(x, xt);
  k_reorder_w   <<<2304, 256, 0, stream>>>(weight, wBt);
  k_reorder_woff<<< 288, 256, 0, stream>>>(w_off, wOffT);
  k_offset_conv <<< 512, 256, 0, stream>>>(xt, wOffT, b_off, offy, offx, maskS);
  k_deform_gemm <<< 512, 256, 0, stream>>>(xt, wBt, offy, offx, maskS,
                                           bias, gamma, beta, rmean, rvar, out);
}
